// PointNet2Seg_SPFE_WSLFA_5583457485368
// MI455X (gfx1250) — matrix operand order repaired and checked by reading, not compiled, not run on hardware
//
#include <hip/hip_runtime.h>
#include <hip/hip_bf16.h>
#include <math.h>

typedef __attribute__((ext_vector_type(16))) __bf16 v16bf;
typedef __attribute__((ext_vector_type(8)))  __bf16 v8bf;
typedef __attribute__((ext_vector_type(8)))  float  v8f;
typedef __attribute__((ext_vector_type(4)))  float  v4f;
typedef __attribute__((ext_vector_type(4)))  unsigned u32x4;
typedef __attribute__((ext_vector_type(8)))  int      i32x8;
typedef __attribute__((ext_vector_type(4)))  int      i32x4;

static __device__ __forceinline__ __bf16 f2bf(float f) { return (__bf16)f; }

#if defined(__HIP_DEVICE_COMPILE__) && __has_builtin(__builtin_amdgcn_global_load_async_to_lds_b32)
#define HAVE_ASYNC_LDS 1
#else
#define HAVE_ASYNC_LDS 0
#endif

#if defined(__HIP_DEVICE_COMPILE__) && __has_builtin(__builtin_amdgcn_tensor_load_to_lds)
#define HAVE_TDM 1
#else
#define HAVE_TDM 0
#endif

#if __has_include(<hip/amd_detail/amd_gfx1250_TDM.h>)
#define TDM_6ARG 1
#else
#define TDM_6ARG 0
#endif

#if HAVE_ASYNC_LDS || HAVE_TDM
#define GPTR(p) ((__attribute__((address_space(1))) int*)(p))
#define LPTR(p) ((__attribute__((address_space(3))) int*)(p))
static __device__ __forceinline__ void wait_async_lds() {
#if __has_builtin(__builtin_amdgcn_s_wait_asynccnt)
  __builtin_amdgcn_s_wait_asynccnt(0);
#else
  asm volatile("s_wait_asynccnt 0" ::: "memory");
#endif
}
static __device__ __forceinline__ void wait_tensorcnt0() {
#if __has_builtin(__builtin_amdgcn_s_wait_tensorcnt)
  __builtin_amdgcn_s_wait_tensorcnt(0);
#else
  asm volatile("s_wait_tensorcnt 0" ::: "memory");
#endif
}
#endif

// ---------------------------------------------------------------------------
// Pack weight sub-matrix (Nout x K, row stride ldw, col offset) into bf16
// with zero-padded row stride Kp (= roundup(K,32)).
// ---------------------------------------------------------------------------
__global__ void pack_w_bf16(const float* __restrict__ W, int ldw, int coloff,
                            int K, int Nout, __bf16* __restrict__ out, int Kp)
{
  size_t total = (size_t)Nout * Kp;
  for (size_t e = blockIdx.x * (size_t)blockDim.x + threadIdx.x; e < total;
       e += (size_t)gridDim.x * blockDim.x) {
    int o = (int)(e / Kp);
    int k = (int)(e % Kp);
    float v = (k < K) ? W[(size_t)o * ldw + coloff + k] : 0.f;
    out[e] = f2bf(v);
  }
}

// ---------------------------------------------------------------------------
// WMMA GEMM:  C(M,N) = A(M,K) * Wp(N,Kp)^T  [+ C if accumulate] [+ bias]
// Block = 128 threads (4 waves) -> 32 x 64 output tile (2 WMMAs per wave/K-step)
// A tile (32x32 f32) staged in LDS via TDM descriptor (tensor_load_to_lds),
// or async-to-LDS, or plain ds stores. OOB handled by TDM zero-fill / clamps.
// ---------------------------------------------------------------------------
__global__ void gemm_wmma_bf16(const float* __restrict__ A, int lda,
                               const __bf16* __restrict__ Wp, int Kp,
                               float* __restrict__ C, int ldc,
                               const float* __restrict__ bias,
                               int Mrows, int Kdim, int Nout, int accumulate)
{
  __shared__ __attribute__((aligned(16))) float As[32 * 36]; // stride 36: conflict-free
  const int tid  = threadIdx.x;        // 0..127
  const int lane = tid & 31;
  const int wv   = tid >> 5;           // wave id 0..3
  const int half = lane >> 4;
  const int l15  = lane & 15;
  const int tm   = blockIdx.x;         // 32-row tile index
  const int rowbase = tm * 32;

  const int bcol  = blockIdx.y * 64 + wv * 16 + l15;
  const int bcolc = (bcol < Nout) ? bcol : (Nout - 1);
  const __bf16* __restrict__ WpRow = Wp + (size_t)bcolc * Kp;

  v8f acc0 = {};
  v8f acc1 = {};
  for (int k0 = 0; k0 < Kp; k0 += 32) {
#if HAVE_TDM
    // --- Tensor Data Mover: one wave DMAs the 32x32 f32 tile into LDS.
    // tensor dims = remaining (rows, cols): TDM zero-fills OOB reads, which
    // handles both the K tail and the M edge. LDS padding 4 dwords per 32
    // dwords reproduces the stride-36 layout.
    if (wv == 0) {
      unsigned ldsb = (unsigned)(unsigned long long)
          (__attribute__((address_space(3))) float*)As;
      unsigned long long ga =
          (unsigned long long)(A + (size_t)rowbase * lda + k0);
      u32x4 g0 = { 1u,                                  // count=1 (valid D#)
                   ldsb,                                // lds_addr
                   (unsigned)ga,                        // global_addr[31:0]
                   ((unsigned)(ga >> 32) & 0x01FFFFFFu) | (2u << 30) }; // +type=2
      int td0 = Kdim - k0;            // remaining cols
      int td1 = Mrows - rowbase;      // remaining rows
      i32x8 g1;
      g1[0] = (2 << 16) | (1 << 20) | (4 << 22) | (3 << 25); // dsz=4B,pad 4/32
      g1[1] = (td0 & 0xFFFF) << 16;                          // tensor_dim0 lo
      g1[2] = ((td0 >> 16) & 0xFFFF) | ((td1 & 0xFFFF) << 16);
      g1[3] = ((td1 >> 16) & 0xFFFF) | (32 << 16);           // tile_dim0=32
      g1[4] = 32;                                            // tile_dim1=32
      g1[5] = lda;                                           // dim0_stride
      g1[6] = 0;
      g1[7] = 0;
      i32x4 z4 = { 0, 0, 0, 0 };
#if TDM_6ARG
      i32x8 z8 = { 0, 0, 0, 0, 0, 0, 0, 0 };
      __builtin_amdgcn_tensor_load_to_lds(g0, g1, z4, z4, z8, 0);
#else
      __builtin_amdgcn_tensor_load_to_lds(g0, g1, z4, z4, 0);
#endif
      wait_tensorcnt0();
    }
#else
    const bool full = (k0 + 32 <= Kdim);
#if HAVE_ASYNC_LDS
    if (full) {
#pragma unroll
      for (int i = 0; i < 8; ++i) {
        int idx = tid + i * 128;          // 0..1023
        int r = idx >> 5, c = idx & 31;
        int row = rowbase + r;
        row = (row < Mrows) ? row : (Mrows - 1);
        __builtin_amdgcn_global_load_async_to_lds_b32(
            GPTR(A + (size_t)row * lda + (k0 + c)), LPTR(&As[r * 36 + c]), 0, 0);
      }
      wait_async_lds();
    } else
#endif
    {
#pragma unroll
      for (int i = 0; i < 8; ++i) {
        int idx = tid + i * 128;
        int r = idx >> 5, c = idx & 31;
        int row = rowbase + r;
        row = (row < Mrows) ? row : (Mrows - 1);
        int k = k0 + c;
        int kc = (k < Kdim) ? k : (Kdim - 1);
        float v = A[(size_t)row * lda + kc];
        As[r * 36 + c] = (k < Kdim) ? v : 0.f;
      }
    }
#endif
    __syncthreads();

    // B fragment: W row bcolc, contiguous K chunk (prepacked bf16, aligned)
    const v8bf* bp = (const v8bf*)(WpRow + k0 + half * 16);
    const v8bf b0 = bp[0];
    const v8bf b1 = bp[1];
    v16bf b = __builtin_shufflevector(b0, b1, 0, 1, 2, 3, 4, 5, 6, 7,
                                      8, 9, 10, 11, 12, 13, 14, 15);

    // A fragments: rows l15 and 16+l15 of the tile;
    // lanes<16 -> K {0..15}, lanes>=16 -> {16..31}
#pragma unroll
    for (int sub = 0; sub < 2; ++sub) {
      const float* ap = As + (sub * 16 + l15) * 36 + half * 16;
      const v4f a0 = *(const v4f*)(ap);
      const v4f a1 = *(const v4f*)(ap + 4);
      const v4f a2 = *(const v4f*)(ap + 8);
      const v4f a3 = *(const v4f*)(ap + 12);
      v16bf a;
#pragma unroll
      for (int i = 0; i < 4; ++i) {
        a[i]      = f2bf(a0[i]);
        a[4 + i]  = f2bf(a1[i]);
        a[8 + i]  = f2bf(a2[i]);
        a[12 + i] = f2bf(a3[i]);
      }
      if (sub == 0)
        acc0 = __builtin_amdgcn_wmma_f32_16x16x32_bf16(false, a, false, b,
                                                       (short)0, acc0, false, false);
      else
        acc1 = __builtin_amdgcn_wmma_f32_16x16x32_bf16(false, a, false, b,
                                                       (short)0, acc1, false, false);
    }
    __syncthreads();
  }

  if (bcol < Nout) {
    const float bia = bias ? bias[bcol] : 0.f;
#pragma unroll
    for (int r = 0; r < 8; ++r) {
      const int m0 = rowbase + half * 8 + r;
      if (m0 < Mrows) {
        const size_t off = (size_t)m0 * ldc + bcol;
        const float v = acc0[r] + bia;
        C[off] = accumulate ? (C[off] + v) : v;
      }
      const int m1 = rowbase + 16 + half * 8 + r;
      if (m1 < Mrows) {
        const size_t off = (size_t)m1 * ldc + bcol;
        const float v = acc1[r] + bia;
        C[off] = accumulate ? (C[off] + v) : v;
      }
    }
  }
}

// ------------------------- per-channel BN stats ----------------------------
__global__ void col_stats(const float* __restrict__ X, int Rows, int Cols,
                          float* __restrict__ mean, float* __restrict__ rstd)
{
  const int c = blockIdx.x;
  __shared__ float ssum[256];
  __shared__ float ssq[256];
  float s = 0.f, q = 0.f;
  for (int r = threadIdx.x; r < Rows; r += blockDim.x) {
    float v = X[(size_t)r * Cols + c];
    s += v; q += v * v;
  }
  ssum[threadIdx.x] = s; ssq[threadIdx.x] = q;
  __syncthreads();
  for (int st = 128; st > 0; st >>= 1) {
    if ((int)threadIdx.x < st) {
      ssum[threadIdx.x] += ssum[threadIdx.x + st];
      ssq[threadIdx.x]  += ssq[threadIdx.x + st];
    }
    __syncthreads();
  }
  if (threadIdx.x == 0) {
    float m = ssum[0] / (float)Rows;
    float v = ssq[0] / (float)Rows - m * m;
    mean[c] = m;
    rstd[c] = rsqrtf(fmaxf(v, 0.f) + 1e-5f);
  }
}

__global__ void bn_relu_inplace(float* __restrict__ X, size_t total, int Cols,
                                const float* __restrict__ mean,
                                const float* __restrict__ rstd)
{
  for (size_t e = blockIdx.x * (size_t)blockDim.x + threadIdx.x; e < total;
       e += (size_t)gridDim.x * blockDim.x) {
    int c = (int)(e % Cols);
    float v = (X[e] - mean[c]) * rstd[c];
    X[e] = fmaxf(v, 0.f);
  }
}

// ------------------------- misc small kernels ------------------------------
__global__ void mean_xyz_k(const float* __restrict__ X, float* __restrict__ out,
                           int B, int N)
{
  const int b = blockIdx.x;
  __shared__ float s[3][256];
  float sx = 0.f, sy = 0.f, sz = 0.f;
  for (int n = threadIdx.x; n < N; n += blockDim.x) {
    const float* p = X + ((size_t)b * N + n) * 3;
    sx += p[0]; sy += p[1]; sz += p[2];
  }
  s[0][threadIdx.x] = sx; s[1][threadIdx.x] = sy; s[2][threadIdx.x] = sz;
  __syncthreads();
  for (int st = 128; st > 0; st >>= 1) {
    if ((int)threadIdx.x < st) {
      s[0][threadIdx.x] += s[0][threadIdx.x + st];
      s[1][threadIdx.x] += s[1][threadIdx.x + st];
      s[2][threadIdx.x] += s[2][threadIdx.x + st];
    }
    __syncthreads();
  }
  if (threadIdx.x < 3) out[b * 3 + threadIdx.x] = s[threadIdx.x][0] / (float)N;
}

__global__ void build_spfe(const float* __restrict__ X,
                           const float* __restrict__ mxyz,
                           float* __restrict__ out, int B, int N)
{
  size_t total = (size_t)B * N;
  for (size_t t = blockIdx.x * (size_t)blockDim.x + threadIdx.x; t < total;
       t += (size_t)gridDim.x * blockDim.x) {
    int b = (int)(t / N);
    const float* p = X + t * 3;
    float* o = out + t * 9;
    o[0] = p[0]; o[1] = p[1]; o[2] = p[2];
    o[3] = p[0] - mxyz[b * 3 + 0];
    o[4] = p[1] - mxyz[b * 3 + 1];
    o[5] = p[2] - mxyz[b * 3 + 2];
    o[6] = 0.f; o[7] = 0.f; o[8] = 0.f;
  }
}

__global__ void gather_centers(const float* __restrict__ xyz_in,
                               float* __restrict__ ctr, int B, int Nin, int M)
{
  int t = blockIdx.x * blockDim.x + threadIdx.x;
  if (t >= B * M) return;
  int b = t / M, m = t % M;
  int src = (int)((float)m * (float)(Nin - 1) / (float)(M - 1));
  if (src > Nin - 1) src = Nin - 1;
  const float* p = xyz_in + ((size_t)b * Nin + src) * 3;
  float* o = ctr + (size_t)t * 3;
  o[0] = p[0]; o[1] = p[1]; o[2] = p[2];
}

// one thread per center: scan Nin points, keep the 32 nearest (sq-dist)
__global__ void knn32(const float* __restrict__ xyz_src,
                      const float* __restrict__ ctr,
                      int* __restrict__ idx_out, int B, int Nin, int M)
{
  int t = blockIdx.x * blockDim.x + threadIdx.x;
  if (t >= B * M) return;
  int b = t / M;
  const float* src = xyz_src + (size_t)b * Nin * 3;
  float cx = ctr[(size_t)t * 3 + 0];
  float cy = ctr[(size_t)t * 3 + 1];
  float cz = ctr[(size_t)t * 3 + 2];
  float best[32]; int bi[32];
  for (int i = 0; i < 32; ++i) { best[i] = 3.4e38f; bi[i] = 0; }
  for (int n = 0; n < Nin; ++n) {
    float dx = src[n * 3 + 0] - cx;
    float dy = src[n * 3 + 1] - cy;
    float dz = src[n * 3 + 2] - cz;
    float d = dx * dx + dy * dy + dz * dz;
    if (d < best[31]) {
      int j = 31;
      while (j > 0 && best[j - 1] > d) { best[j] = best[j - 1]; bi[j] = bi[j - 1]; --j; }
      best[j] = d; bi[j] = n;
    }
  }
  int* o = idx_out + (size_t)t * 32;
  for (int i = 0; i < 32; ++i) o[i] = bi[i];
}

// cat[b,m,k,:] = [ xyz[n]-ctr[m] (3) , feat[n] (Cf) ]
__global__ void build_cat(const float* __restrict__ xyz_src,
                          const float* __restrict__ f_src,
                          const float* __restrict__ ctr,
                          const int* __restrict__ knn,
                          float* __restrict__ cat,
                          int B, int Nin, int M, int Cf)
{
  const int Ccat = 3 + Cf;
  size_t total = (size_t)B * M * 32 * Ccat;
  for (size_t e = blockIdx.x * (size_t)blockDim.x + threadIdx.x; e < total;
       e += (size_t)gridDim.x * blockDim.x) {
    size_t row = e / Ccat;
    int col = (int)(e % Ccat);
    size_t bm = row >> 5;                 // b*M + m
    int b = (int)(bm / M);
    int n = knn[row];
    float v;
    if (col < 3)
      v = xyz_src[((size_t)b * Nin + n) * 3 + col] - ctr[bm * 3 + col];
    else
      v = f_src[((size_t)b * Nin + n) * Cf + (col - 3)];
    cat[e] = v;
  }
}

__global__ void mean_over_k(const float* __restrict__ fp, float* __restrict__ fpm,
                            int BM, int Cout)
{
  size_t total = (size_t)BM * Cout;
  for (size_t t = blockIdx.x * (size_t)blockDim.x + threadIdx.x; t < total;
       t += (size_t)gridDim.x * blockDim.x) {
    size_t bm = t / Cout;
    int o = (int)(t % Cout);
    const float* p = fp + (bm * 32) * (size_t)Cout + o;
    float s = 0.f;
    for (int k = 0; k < 32; ++k) s += p[(size_t)k * Cout];
    fpm[t] = s * (1.f / 32.f);
  }
}

__global__ void sub_bcast(float* __restrict__ alpha, const float* __restrict__ v,
                          int BM, int Cout)
{
  size_t total = (size_t)BM * 32 * Cout;
  for (size_t e = blockIdx.x * (size_t)blockDim.x + threadIdx.x; e < total;
       e += (size_t)gridDim.x * blockDim.x) {
    int o = (int)(e % Cout);
    size_t bm = e / ((size_t)32 * Cout);
    alpha[e] -= v[bm * Cout + o];
  }
}

// per (b,m,o): softmax over k of alpha, then weighted sum of fp
__global__ void softmax_wsum(const float* __restrict__ alpha,
                             const float* __restrict__ fp,
                             float* __restrict__ out, int BM, int Cout)
{
  size_t total = (size_t)BM * Cout;
  for (size_t t = blockIdx.x * (size_t)blockDim.x + threadIdx.x; t < total;
       t += (size_t)gridDim.x * blockDim.x) {
    size_t bm = t / Cout;
    int o = (int)(t % Cout);
    const float* a = alpha + (bm * 32) * (size_t)Cout + o;
    const float* f = fp    + (bm * 32) * (size_t)Cout + o;
    float mx = -3.4e38f;
    for (int k = 0; k < 32; ++k) mx = fmaxf(mx, a[(size_t)k * Cout]);
    float s = 0.f, acc = 0.f;
    for (int k = 0; k < 32; ++k) {
      float e = __expf(a[(size_t)k * Cout] - mx);
      s += e;
      acc += e * f[(size_t)k * Cout];
    }
    out[t] = acc / s;
  }
}

// 3-NN + inverse-distance weights for feature propagation
__global__ void knn3_w(const float* __restrict__ xyz_l,
                       const float* __restrict__ xyz_h,
                       int* __restrict__ idx3, float* __restrict__ w3,
                       int B, int Nl, int Nh)
{
  int t = blockIdx.x * blockDim.x + threadIdx.x;
  if (t >= B * Nl) return;
  int b = t / Nl;
  const float* src = xyz_h + (size_t)b * Nh * 3;
  float cx = xyz_l[(size_t)t * 3 + 0];
  float cy = xyz_l[(size_t)t * 3 + 1];
  float cz = xyz_l[(size_t)t * 3 + 2];
  float d0 = 3.4e38f, d1 = 3.4e38f, d2 = 3.4e38f;
  int i0 = 0, i1 = 0, i2 = 0;
  for (int n = 0; n < Nh; ++n) {
    float dx = src[n * 3 + 0] - cx;
    float dy = src[n * 3 + 1] - cy;
    float dz = src[n * 3 + 2] - cz;
    float d = dx * dx + dy * dy + dz * dz;
    if (d < d0)      { d2 = d1; i2 = i1; d1 = d0; i1 = i0; d0 = d; i0 = n; }
    else if (d < d1) { d2 = d1; i2 = i1; d1 = d; i1 = n; }
    else if (d < d2) { d2 = d; i2 = n; }
  }
  float w0 = 1.f / fmaxf(sqrtf(fmaxf(d0, 0.f)), 1e-8f);
  float w1 = 1.f / fmaxf(sqrtf(fmaxf(d1, 0.f)), 1e-8f);
  float w2 = 1.f / fmaxf(sqrtf(fmaxf(d2, 0.f)), 1e-8f);
  float s = w0 + w1 + w2;
  idx3[(size_t)t * 3 + 0] = i0; idx3[(size_t)t * 3 + 1] = i1; idx3[(size_t)t * 3 + 2] = i2;
  w3[(size_t)t * 3 + 0] = w0 / s; w3[(size_t)t * 3 + 1] = w1 / s; w3[(size_t)t * 3 + 2] = w2 / s;
}

// xcat[b,n,:] = [ interp(fh) (Ch) , fl (Cl) ]
__global__ void build_fpcat(const float* __restrict__ fh,
                            const float* __restrict__ fl,
                            const int* __restrict__ idx3,
                            const float* __restrict__ w3,
                            float* __restrict__ xcat,
                            int B, int Nl, int Nh, int Ch, int Cl)
{
  const int Cc = Ch + Cl;
  size_t total = (size_t)B * Nl * Cc;
  for (size_t e = blockIdx.x * (size_t)blockDim.x + threadIdx.x; e < total;
       e += (size_t)gridDim.x * blockDim.x) {
    size_t row = e / Cc;
    int c = (int)(e % Cc);
    int b = (int)(row / Nl);
    float v;
    if (c < Ch) {
      const int*   id = idx3 + row * 3;
      const float* w  = w3 + row * 3;
      const float* base = fh + (size_t)b * Nh * Ch + c;
      v = w[0] * base[(size_t)id[0] * Ch]
        + w[1] * base[(size_t)id[1] * Ch]
        + w[2] * base[(size_t)id[2] * Ch];
    } else {
      v = fl[row * (size_t)Cl + (c - Ch)];
    }
    xcat[e] = v;
  }
}

// ---------------------------------------------------------------------------
extern "C" void kernel_launch(void* const* d_in, const int* in_sizes, int n_in,
                              void* d_out, int out_size, void* d_ws, size_t ws_size,
                              hipStream_t stream)
{
  (void)in_sizes; (void)n_in; (void)out_size; (void)ws_size;

  const int B = 8, N = 4096;
  const int M1 = 1024, M2 = 512, M3 = 256;

  const float* X       = (const float*)d_in[0];
  const float* W_spfe1 = (const float*)d_in[1];
  const float* W_spfe2 = (const float*)d_in[3];
  const float* Wf1 = (const float*)d_in[5];
  const float* Wa1 = (const float*)d_in[7];
  const float* Wf2 = (const float*)d_in[9];
  const float* Wa2 = (const float*)d_in[11];
  const float* Wf3 = (const float*)d_in[13];
  const float* Wa3 = (const float*)d_in[15];
  const float* Wfp3 = (const float*)d_in[17];
  const float* Wfp2 = (const float*)d_in[19];
  const float* Wfp1 = (const float*)d_in[21];
  const float* Wh1  = (const float*)d_in[23];
  const float* Wh2  = (const float*)d_in[25];
  const float* bh2  = (const float*)d_in[26];
  float* out = (float*)d_out;

  // ---- workspace bump allocator (floats) ----
  float* wsf = (float*)d_ws;
  size_t off = 0;
  auto alloc = [&](size_t n) -> float* {
    float* p = wsf + off;
    off += (n + 63) & ~(size_t)63;
    return p;
  };
  float* meanxyz  = alloc((size_t)B * 3);
  float* spfe_in  = alloc((size_t)B * N * 9);
  float* h0       = alloc((size_t)B * N * 64);
  float* f0       = alloc((size_t)B * N * 64);
  float* stat_m   = alloc(1024);
  float* stat_r   = alloc(1024);
  float* xyz1 = alloc((size_t)B * M1 * 3);
  float* xyz2 = alloc((size_t)B * M2 * 3);
  float* xyz3 = alloc((size_t)B * M3 * 3);
  float* f1   = alloc((size_t)B * M1 * 128);
  float* f2   = alloc((size_t)B * M2 * 256);
  float* f3   = alloc((size_t)B * M3 * 512);
  float* fup2 = alloc((size_t)B * M2 * 256);
  float* fup1 = alloc((size_t)B * M1 * 128);
  float* fup0 = alloc((size_t)B * N * 128);
  float* hh   = alloc((size_t)B * N * 128);
  int*   knn  = (int*)alloc((size_t)B * M1 * 32);
  int*   idx3 = (int*)alloc((size_t)B * N * 3);
  float* w3   = alloc((size_t)B * N * 3);
  __bf16* wpack = (__bf16*)alloc(131072);          // 262144 bf16: max packed weight
  float* catb = alloc((size_t)B * M1 * 32 * 67);   // max cat / xcat region
  float* fpb  = alloc((size_t)B * M1 * 32 * 128);  // max fp region
  float* alfb = alloc((size_t)B * M1 * 32 * 128);  // max alpha region
  float* fpmb  = alloc((size_t)B * M1 * 128);
  float* fpmWb = alloc((size_t)B * M1 * 128);

  auto gsz = [](size_t tot) { return dim3((unsigned)((tot + 255) / 256)); };
  auto KP  = [](int k) { return (k + 31) & ~31; };

  auto launch_pack = [&](const float* W, int ldw, int coloff, int K, int Nn) {
    int Kp = KP(K);
    size_t tot = (size_t)Nn * Kp;
    pack_w_bf16<<<gsz(tot), 256, 0, stream>>>(W, ldw, coloff, K, Nn, wpack, Kp);
  };
  auto launch_gemm = [&](const float* A, int lda, float* C, int ldc,
                         const float* bias, int M, int K, int Nn, int acc) {
    dim3 g((M + 31) / 32, (Nn + 63) / 64);
    gemm_wmma_bf16<<<g, 128, 0, stream>>>(A, lda, wpack, KP(K), C, ldc, bias,
                                          M, K, Nn, acc);
  };
  auto launch_bn = [&](float* Xb, int R, int C) {
    col_stats<<<C, 256, 0, stream>>>(Xb, R, C, stat_m, stat_r);
    size_t tot = (size_t)R * C;
    bn_relu_inplace<<<gsz(tot), 256, 0, stream>>>(Xb, tot, C, stat_m, stat_r);
  };

  // ---- SPFE ----
  mean_xyz_k<<<B, 256, 0, stream>>>(X, meanxyz, B, N);
  build_spfe<<<gsz((size_t)B * N), 256, 0, stream>>>(X, meanxyz, spfe_in, B, N);
  launch_pack(W_spfe1, 9, 0, 9, 64);
  launch_gemm(spfe_in, 9, h0, 64, nullptr, B * N, 9, 64, 0);
  launch_bn(h0, B * N, 64);
  launch_pack(W_spfe2, 64, 0, 64, 64);
  launch_gemm(h0, 64, f0, 64, nullptr, B * N, 64, 64, 0);
  launch_bn(f0, B * N, 64);

  // ---- SA layer (WSLFA) ----
  auto sa_layer = [&](const float* xyz_in, int Nin, const float* f_in, int Cf,
                      int M, int Cout, const float* Wf, const float* Wa,
                      float* xyz_out, float* f_out) {
    const int Ccat = 3 + Cf;
    const int Ca = Ccat + Cout;    // ain channels
    const int BM = B * M;
    const size_t R = (size_t)BM * 32;
    gather_centers<<<gsz((size_t)BM), 256, 0, stream>>>(xyz_in, xyz_out, B, Nin, M);
    knn32<<<(BM + 63) / 64, 64, 0, stream>>>(xyz_in, xyz_out, knn, B, Nin, M);
    build_cat<<<gsz(R * Ccat), 256, 0, stream>>>(xyz_in, f_in, xyz_out, knn,
                                                 catb, B, Nin, M, Cf);
    // fp = relu(bn(cat * Wf^T))
    launch_pack(Wf, Ccat, 0, Ccat, Cout);
    launch_gemm(catb, Ccat, fpb, Cout, nullptr, (int)R, Ccat, Cout, 0);
    launch_bn(fpb, (int)R, Cout);
    mean_over_k<<<gsz((size_t)BM * Cout), 256, 0, stream>>>(fpb, fpmb, BM, Cout);
    // alpha_pre = cat*Wa1^T + fp*Wa2^T - bcast(fpm*Wa2^T)
    launch_pack(Wa, Ca, 0, Ccat, Cout);
    launch_gemm(catb, Ccat, alfb, Cout, nullptr, (int)R, Ccat, Cout, 0);
    launch_pack(Wa, Ca, Ccat, Cout, Cout);
    launch_gemm(fpb, Cout, alfb, Cout, nullptr, (int)R, Cout, Cout, 1);
    launch_gemm(fpmb, Cout, fpmWb, Cout, nullptr, BM, Cout, Cout, 0);
    sub_bcast<<<gsz(R * Cout), 256, 0, stream>>>(alfb, fpmWb, BM, Cout);
    launch_bn(alfb, (int)R, Cout);
    softmax_wsum<<<gsz((size_t)BM * Cout), 256, 0, stream>>>(alfb, fpb, f_out, BM, Cout);
  };

  sa_layer(X,    N,  f0, 64,  M1, 128, Wf1, Wa1, xyz1, f1);
  sa_layer(xyz1, M1, f1, 128, M2, 256, Wf2, Wa2, xyz2, f2);
  sa_layer(xyz2, M2, f2, 256, M3, 512, Wf3, Wa3, xyz3, f3);

  // ---- FP layers ----
  auto fp_layer = [&](const float* xyz_l, int Nl, const float* xyz_h, int Nh,
                      const float* f_l, int Cl, const float* f_h, int Ch,
                      const float* W, float* o, int Co) {
    const int BNl = B * Nl;
    knn3_w<<<(BNl + 127) / 128, 128, 0, stream>>>(xyz_l, xyz_h, idx3, w3, B, Nl, Nh);
    build_fpcat<<<gsz((size_t)BNl * (Ch + Cl)), 256, 0, stream>>>(
        f_h, f_l, idx3, w3, catb, B, Nl, Nh, Ch, Cl);
    launch_pack(W, Ch + Cl, 0, Ch + Cl, Co);
    launch_gemm(catb, Ch + Cl, o, Co, nullptr, BNl, Ch + Cl, Co, 0);
    launch_bn(o, BNl, Co);
  };

  fp_layer(xyz2, M2, xyz3, M3, f2, 256, f3,   512, Wfp3, fup2, 256);
  fp_layer(xyz1, M1, xyz2, M2, f1, 128, fup2, 256, Wfp2, fup1, 128);
  fp_layer(X,    N,  xyz1, M1, f0, 64,  fup1, 128, Wfp1, fup0, 128);

  // ---- head ----
  launch_pack(Wh1, 128, 0, 128, 128);
  launch_gemm(fup0, 128, hh, 128, nullptr, B * N, 128, 128, 0);
  launch_bn(hh, B * N, 128);
  launch_pack(Wh2, 128, 0, 128, 50);
  launch_gemm(hh, 128, out, 50, bh2, B * N, 128, 50, 0);
}
